// MoE_47055661695574
// MI455X (gfx1250) — matrix operand order repaired and checked by reading, not compiled, not run on hardware
//
#include <hip/hip_runtime.h>

typedef __attribute__((ext_vector_type(16))) __bf16       v16bf;
typedef __attribute__((ext_vector_type(8)))  float        v8f;
typedef __attribute__((ext_vector_type(8)))  unsigned int v8u;

#define N_TOKENS 32768
#define D_MODEL  1024
#define BM 128
#define BN 128
#define KSTEPS (D_MODEL / 32)   // 32 k-steps of K=32
#define NTILES (D_MODEL / 16)   // 64 column tiles of 16

// ---------------------------------------------------------------------------
// Prepack weights into bf16 WMMA B-fragment register images in d_ws (4 MB):
//   slot e=0 : bf16(W1)
//   slot e=1 : bf16(W2 - W1)   (W-difference trick: y = x@W1 + (x*m1)@Wd)
// One block per (e, n-tile nt, k-step ks): 32x16 bf16 tile = 256 dwords.
// Lane L of the GEMM wave owns dwords [L*8 .. L*8+7] (two global_load_b128).
// B order used here (contiguous k, same as A): lanes 0-15: VGPR j = {K=2j+1 | K=2j} at
// N = lane; lanes 16-31: K += 16, N = lane-16. (low 16 bits = even K)
// ---------------------------------------------------------------------------
__global__ __launch_bounds__(256) void prepack_w(const float* __restrict__ W1,
                                                 const float* __restrict__ W2,
                                                 unsigned int* __restrict__ wp) {
  unsigned int flat = blockIdx.x * 256u + threadIdx.x;  // < 2*64*32*256
  unsigned int d    = flat & 255u;   // dword within block
  unsigned int blk  = flat >> 8;
  unsigned int ks   = blk & 31u;
  unsigned int nt   = (blk >> 5) & 63u;
  unsigned int e    = blk >> 11;
  unsigned int lane = d >> 3;
  unsigned int j    = d & 7u;
  unsigned int n    = nt * 16u + (lane & 15u);
  unsigned int k    = ks * 32u + (lane >> 4) * 16u + 2u * j;
  size_t i0 = (size_t)k * D_MODEL + n;
  size_t i1 = i0 + D_MODEL;  // (k+1, n)
  float lo = e ? (W2[i0] - W1[i0]) : W1[i0];
  float hi = e ? (W2[i1] - W1[i1]) : W1[i1];
  union { unsigned int u; __bf16 h[2]; } p;
  p.h[0] = (__bf16)lo;   // low 16 bits = even K
  p.h[1] = (__bf16)hi;
  wp[flat] = p.u;
}

// ---------------------------------------------------------------------------
// MoE dual-expert GEMM via W-difference:
//   out = x@W1 + (x*m1)@(W2-W1) + (route ? b2 : b1)
// 256 threads = 8 waves; block tile 128x128; wave tile 32 rows x 64 cols
// (2x4 accumulator tiles). Per k-step per tile: two chained
// v_wmma_f32_16x16x32_bf16 (dense A x W1, route-masked A x Wd).
// ---------------------------------------------------------------------------
__global__ __launch_bounds__(256) void moe_gemm(const float* __restrict__ x,
                                                const unsigned int* __restrict__ wp,
                                                const float* __restrict__ b1,
                                                const float* __restrict__ b2,
                                                const int* __restrict__ route,
                                                float* __restrict__ out) {
  const int lane   = threadIdx.x & 31;
  const int wave   = threadIdx.x >> 5;
  const int wave_m = wave & 3;    // 4 waves down M
  const int wave_n = wave >> 2;   // 2 waves across N
  const int base_m  = blockIdx.y * BM + wave_m * 32;
  const int base_nt = blockIdx.x * (BN / 16) + wave_n * 4;  // global 16-col tile

  const int khalf = lane >> 4;             // 0: lanes 0-15, 1: lanes 16-31
  const int mrow0 = base_m + (lane & 15);  // A-frag row, m-tile 0
  const int mrow1 = mrow0 + 16;            // A-frag row, m-tile 1
  // route: 0 -> expert1 (W1 only), nonzero -> expert2 (add Wd term).
  const bool sel0 = route[mrow0] != 0;     // k-loop-invariant VCC
  const bool sel1 = route[mrow1] != 0;

  v8f acc[2][4];
#pragma unroll
  for (int mt = 0; mt < 2; ++mt)
#pragma unroll
    for (int nt = 0; nt < 4; ++nt) acc[mt][nt] = (v8f)0.0f;

  for (int ks = 0; ks < KSTEPS; ++ks) {
    const int k0 = ks * 32;

    // A fragments. Layout: lanes 0-15 hold K 0-15 (VGPR0-7);
    // lanes 16-31 hold K 16-31; M = lane & 15.
    // af = bf16(x) (dense; elementwise writes fuse to v_cvt_pk_bf16_f32),
    // am = route-masked copy (v_cndmask_b32 on the packed dwords).
    v16bf af[2], am[2];
#pragma unroll
    for (int mt = 0; mt < 2; ++mt) {
      const int  row = mt ? mrow1 : mrow0;
      const bool sel = mt ? sel1 : sel0;
      const float* __restrict__ px = x + (size_t)row * D_MODEL + k0 + khalf * 16;
      v8f g0 = *reinterpret_cast<const v8f*>(px);       // K rel 0..7  (this half)
      v8f g1 = *reinterpret_cast<const v8f*>(px + 8);   // K rel 8..15 (this half)
      v16bf f;
#pragma unroll
      for (int i = 0; i < 8; ++i) {
        f[i]     = (__bf16)g0[i];
        f[i + 8] = (__bf16)g1[i];
      }
      v8u dw = __builtin_bit_cast(v8u, f);
      v8u dm;
#pragma unroll
      for (int i = 0; i < 8; ++i) dm[i] = sel ? dw[i] : 0u;  // v_cndmask_b32
      af[mt] = f;
      am[mt] = __builtin_bit_cast(v16bf, dm);
    }

    // Per 16-col tile: load both prepacked fragments (W1, Wd) and issue two
    // chained WMMAs per accumulator tile.
#pragma unroll
    for (int nt = 0; nt < 4; ++nt) {
      const int ntg = base_nt + nt;
      const unsigned int* p0 =
          wp + ((size_t)(0 * NTILES + ntg) * KSTEPS + ks) * 256 + lane * 8;
      const unsigned int* p1 =
          wp + ((size_t)(1 * NTILES + ntg) * KSTEPS + ks) * 256 + lane * 8;
      v16bf bw1 = __builtin_bit_cast(v16bf, *reinterpret_cast<const v8u*>(p0));
      v16bf bwd = __builtin_bit_cast(v16bf, *reinterpret_cast<const v8u*>(p1));
#pragma unroll
      for (int mt = 0; mt < 2; ++mt) {
        acc[mt][nt] = __builtin_amdgcn_wmma_f32_16x16x32_bf16(
            false, af[mt], false, bw1, (short)0, acc[mt][nt], false, false);
        acc[mt][nt] = __builtin_amdgcn_wmma_f32_16x16x32_bf16(
            false, am[mt], false, bwd, (short)0, acc[mt][nt], false, false);
      }
    }
  }

  // Epilogue: routed bias + store. C/D layout: lanes 0-15: N = lane,
  // VGPR v -> M = v; lanes 16-31: N = lane-16, M = 8+v.
#pragma unroll
  for (int mt = 0; mt < 2; ++mt) {
    int rr[8];
#pragma unroll
    for (int v = 0; v < 8; ++v)
      rr[v] = route[base_m + mt * 16 + khalf * 8 + v];
#pragma unroll
    for (int nt = 0; nt < 4; ++nt) {
      const int n = (base_nt + nt) * 16 + (lane & 15);
      const float bia0 = b1[n];
      const float bia1 = b2[n];
#pragma unroll
      for (int v = 0; v < 8; ++v) {
        const int row = base_m + mt * 16 + khalf * 8 + v;
        out[(size_t)row * D_MODEL + n] = acc[mt][nt][v] + (rr[v] ? bia1 : bia0);
      }
    }
  }
}

extern "C" void kernel_launch(void* const* d_in, const int* in_sizes, int n_in,
                              void* d_out, int out_size, void* d_ws, size_t ws_size,
                              hipStream_t stream) {
  // setup_inputs order: x, W1, b1, W2, b2, route
  const float* x     = (const float*)d_in[0];
  const float* W1    = (const float*)d_in[1];
  const float* b1    = (const float*)d_in[2];
  const float* W2    = (const float*)d_in[3];
  const float* b2    = (const float*)d_in[4];
  const int*   route = (const int*)d_in[5];
  float* out = (float*)d_out;
  unsigned int* wp = (unsigned int*)d_ws;  // needs 2*64*32*1024 B = 4 MB

  // Pack W1 and (W2-W1) into WMMA B-fragment images (bf16), L2-resident.
  prepack_w<<<dim3(2 * NTILES * KSTEPS), dim3(256), 0, stream>>>(W1, W2, wp);
  // Fused dual-expert GEMM. N-blocks innermost for L2 reuse of x tiles.
  moe_gemm<<<dim3(D_MODEL / BN, N_TOKENS / BM), dim3(256), 0, stream>>>(
      x, wp, b1, b2, route, out);
}
